// QThreeLayer_3461743640594
// MI455X (gfx1250) — 8-bit matrix operand order repaired and checked by reading, compiled, not run on hardware
//
#include <hip/hip_runtime.h>


typedef __attribute__((ext_vector_type(8))) int v8i;

__device__ __forceinline__ int quant6(float x, float inv_sf) {
    float r = rintf(x * inv_sf);
    r = fminf(fmaxf(r, -32.0f), 31.0f);
    return (int)r;
}

__device__ __forceinline__ unsigned pack4(int a, int b, int c, int d) {
    unsigned lo = __builtin_amdgcn_perm((unsigned)b, (unsigned)a, 0x0C0C0400u);
    unsigned hi = __builtin_amdgcn_perm((unsigned)d, (unsigned)c, 0x04000C0Cu);
    return lo | hi;
}

__global__ void prep_weights(const float* __restrict__ W1, const float* __restrict__ W2,
                             const float* __restrict__ W3, const float* __restrict__ W4,
                             float* __restrict__ wsc1, float* __restrict__ wsc2,
                             float* __restrict__ wsc3, float* __restrict__ wsc4,
                             unsigned* __restrict__ pw1, unsigned* __restrict__ pw2,
                             unsigned* __restrict__ pw3, unsigned* __restrict__ pw4) {
    int tid = threadIdx.x;
    for (int i = tid; i < 144; i += 256) {
        const float* W; float* wsc; int IN, OUT, n;
        if (i < 64)       { W = W1; wsc = wsc1; IN = 16; OUT = 64; n = i; }
        else if (i < 96)  { W = W2; wsc = wsc2; IN = 64; OUT = 32; n = i - 64; }
        else if (i < 128) { W = W3; wsc = wsc3; IN = 32; OUT = 32; n = i - 96; }
        else              { W = W4; wsc = wsc4; IN = 32; OUT = 5;  n = i - 128; }
        float s = 1.0f;
        if (n < OUT) {
            float m = 0.0f;
            for (int k = 0; k < IN; ++k) m = fmaxf(m, fabsf(W[n * IN + k]));
            s = fmaxf(m, 1e-8f) / 31.0f;
        }
        wsc[n] = s;
    }
    __syncthreads();
    for (int t = tid; t < 2304; t += 256) {
        const float* W; const float* wsc; unsigned* pw; int IN, OUT, local;
        if (t < 1024)      { W = W1; wsc = wsc1; pw = pw1; IN = 16; OUT = 64; local = t; }
        else if (t < 1536) { W = W2; wsc = wsc2; pw = pw2; IN = 64; OUT = 32; local = t - 1024; }
        else if (t < 2048) { W = W3; wsc = wsc3; pw = pw3; IN = 32; OUT = 32; local = t - 1536; }
        else               { W = W4; wsc = wsc4; pw = pw4; IN = 32; OUT = 5;  local = t - 2048; }
        int tile = local >> 8;
        int rem  = local & 255;
        int v    = rem >> 5;
        int lane = rem & 31;
        int n    = tile * 16 + (lane & 15);
        int half = lane >> 4;
        int kbase = ((v >> 2) << 5) + (half << 4) + ((v & 3) << 2);
        unsigned word = 0;
        if (n < OUT) {
            float inv = 1.0f / wsc[n];
            int q[4];
#pragma unroll
            for (int j = 0; j < 4; ++j) {
                int k = kbase + j;
                q[j] = (k < IN) ? quant6(W[n * IN + k], inv) : 0;
            }
            word = pack4(q[0], q[1], q[2], q[3]);
        }
        pw[local] = word;
    }
}

__global__ void absmax_f32(const float* __restrict__ x, long long n4,
                           unsigned* __restrict__ out) {
    long long i = (long long)blockIdx.x * blockDim.x + threadIdx.x;
    long long stride = (long long)gridDim.x * blockDim.x;
    const float4* x4 = (const float4*)x;
    float m = 0.0f;
    for (; i < n4; i += stride) {
        float4 v = x4[i];
        m = fmaxf(m, fmaxf(fmaxf(fabsf(v.x), fabsf(v.y)), fmaxf(fabsf(v.z), fabsf(v.w))));
    }
    __shared__ float sred[256];
    int tid = threadIdx.x;
    sred[tid] = m;
    __syncthreads();
    for (int s = 128; s > 0; s >>= 1) {
        if (tid < s) sred[tid] = fmaxf(sred[tid], sred[tid + s]);
        __syncthreads();
    }
    if (tid == 0) atomicMax(out, __float_as_uint(sred[0]));
}

template <int IN, bool IN_IS_F32>
__device__ __forceinline__ v8i pack_A(const void* __restrict__ in_, long long M, int half,
                                      float inv_sf) {
    v8i A = {0, 0, 0, 0, 0, 0, 0, 0};
    if constexpr (IN_IS_F32) {
        const float* x = (const float*)in_ + M * IN;
        if (half == 0) {
#pragma unroll
        for (int v = 0; v < 4; ++v) {
        float4 a = *(const float4*)(x + v * 4);
        A[v] = (int)pack4(quant6(a.x, inv_sf), quant6(a.y, inv_sf),
                          quant6(a.z, inv_sf), quant6(a.w, inv_sf));
        }
        }
    } else {
        const _Float16* x = (const _Float16*)in_ + M * IN;
#pragma unroll
        for (int v = 0; v < 8; ++v) {
            int kbase = ((v >> 2) << 5) + (half << 4) + ((v & 3) << 2);
            if (kbase < IN) {
                union { uint2 u; _Float16 h[4]; } d;
                d.u = *(const uint2*)(x + kbase);
                A[v] = (int)pack4(quant6((float)d.h[0], inv_sf), quant6((float)d.h[1], inv_sf),
                                  quant6((float)d.h[2], inv_sf), quant6((float)d.h[3], inv_sf));
            }
        }
    }
    return A;
}

template <int IN, int OUT, bool IN_IS_F32, int MT>
__global__ __launch_bounds__(256) void qlinear_relu(
        const void* __restrict__ in_, const float* __restrict__ bias,
        const float* __restrict__ wsc, const unsigned* __restrict__ pw,
        const unsigned* __restrict__ max_in, unsigned* __restrict__ max_out,
        _Float16* __restrict__ out, int nrows) {
    constexpr int NT = (OUT + 15) / 16;
    int tid  = threadIdx.x;
    int lane = tid & 31;
    int wave = tid >> 5;
    int half = lane >> 4;
    int l15  = lane & 15;
    long long M0 = ((long long)blockIdx.x * 8 + wave) * (MT * 16);
    float lmax = 0.0f;

    if (M0 < nrows) {
        float sf = fmaxf(__uint_as_float(*max_in), 1e-8f) / 31.0f;
        float inv_sf = 1.0f / sf;

        v8i Bv[NT];
        float bsf[NT], bi[NT];
#pragma unroll
        for (int t = 0; t < NT; ++t) {
#pragma unroll
            for (int v = 0; v < 8; ++v) Bv[t][v] = (int)pw[(t * 8 + v) * 32 + lane];
            int N = t * 16 + l15;
            bsf[t] = wsc[N] * sf;
            bi[t]  = fminf(fmaxf(rintf(bias[N] / bsf[t]), -32.0f), 31.0f);
        }

#pragma unroll
        for (int mt = 0; mt < MT; ++mt) {
            long long Mt = M0 + mt * 16;
            if (Mt >= nrows) break;
            v8i A = pack_A<IN, IN_IS_F32>(in_, Mt + l15, half, inv_sf);

            v8i acc[NT];
#pragma unroll
            for (int t = 0; t < NT; ++t) {
                v8i c = {0, 0, 0, 0, 0, 0, 0, 0};
                acc[t] = __builtin_amdgcn_wmma_i32_16x16x64_iu8(true, A, true, Bv[t], c,
                                                                false, false);
            }
#pragma unroll
            for (int t = 0; t < NT; ++t) {
                int N = t * 16 + l15;
#pragma unroll
                for (int r = 0; r < 8; ++r) {
                    long long Mo = Mt + half * 8 + r;
                    float val = ((float)acc[t][r] + bi[t]) * bsf[t];
                    val = fmaxf(val, 0.0f);
                    lmax = fmaxf(lmax, val);
                    out[Mo * OUT + N] = (_Float16)val;
                }
            }
        }
    }

    __shared__ float sred[256];
    sred[tid] = lmax;
    __syncthreads();
    for (int s = 128; s > 0; s >>= 1) {
        if (tid < s) sred[tid] = fmaxf(sred[tid], sred[tid + s]);
        __syncthreads();
    }
    if (tid == 0) atomicMax(max_out, __float_as_uint(sred[0]));
}

template <int MT>
__global__ __launch_bounds__(256) void qlinear4_softmax(
        const _Float16* __restrict__ in_, const float* __restrict__ bias,
        const float* __restrict__ wsc, const unsigned* __restrict__ pw,
        const unsigned* __restrict__ max_in, float* __restrict__ out, int nrows) {
    constexpr int RPB = 8 * MT * 16;
    __shared__ float slog[RPB][5];
    int tid  = threadIdx.x;
    int lane = tid & 31;
    int wave = tid >> 5;
    int half = lane >> 4;
    int l15  = lane & 15;
    long long M0 = ((long long)blockIdx.x * 8 + wave) * (MT * 16);

    if (M0 < nrows) {
        float sf = fmaxf(__uint_as_float(*max_in), 1e-8f) / 31.0f;
        float inv_sf = 1.0f / sf;

        v8i Bv = {0, 0, 0, 0, 0, 0, 0, 0};
#pragma unroll
        for (int v = 0; v < 8; ++v) Bv[v] = (int)pw[v * 32 + lane];
        int N = l15;
        float bsf = wsc[N] * sf;
        float bi  = fminf(fmaxf(rintf(bias[N] / bsf), -32.0f), 31.0f);

#pragma unroll
        for (int mt = 0; mt < MT; ++mt) {
            long long Mt = M0 + mt * 16;
            if (Mt >= nrows) break;
            v8i A = pack_A<32, false>((const void*)in_, Mt + l15, half, inv_sf);
            v8i c = {0, 0, 0, 0, 0, 0, 0, 0};
            v8i acc = __builtin_amdgcn_wmma_i32_16x16x64_iu8(true, A, true, Bv, c,
                                                             false, false);
            if (N < 5) {
                int rbase = wave * (MT * 16) + mt * 16 + half * 8;
#pragma unroll
                for (int r = 0; r < 8; ++r)
                    slog[rbase + r][N] = ((float)acc[r] + bi) * bsf;
            }
        }
    }
    __syncthreads();

    for (int rr = tid; rr < RPB; rr += 256) {
        long long row = (long long)blockIdx.x * RPB + rr;
        if (row < nrows) {
            float v[5], m = -3.402823466e+38f;
#pragma unroll
            for (int i = 0; i < 5; ++i) { v[i] = slog[rr][i]; m = fmaxf(m, v[i]); }
            float s = 0.0f;
#pragma unroll
            for (int i = 0; i < 5; ++i) { v[i] = expf(v[i] - m); s += v[i]; }
            float inv = 1.0f / s;
#pragma unroll
            for (int i = 0; i < 5; ++i) out[row * 5 + i] = v[i] * inv;
        }
    }
}

extern "C" void kernel_launch(void* const* d_in, const int* in_sizes, int n_in,
                              void* d_out, int out_size, void* d_ws, size_t ws_size,
                              hipStream_t stream) {
    const float* x  = (const float*)d_in[0];
    const float* W1 = (const float*)d_in[1];
    const float* b1 = (const float*)d_in[2];
    const float* W2 = (const float*)d_in[3];
    const float* b2 = (const float*)d_in[4];
    const float* W3 = (const float*)d_in[5];
    const float* b3 = (const float*)d_in[6];
    const float* W4 = (const float*)d_in[7];
    const float* b4 = (const float*)d_in[8];

    int nrows = in_sizes[0] / 16;

    char* ws = (char*)d_ws;
    unsigned* maxbits = (unsigned*)ws;
    float*    wsc1 = (float*)(ws + 256);
    float*    wsc2 = (float*)(ws + 512);
    float*    wsc3 = (float*)(ws + 640);
    float*    wsc4 = (float*)(ws + 768);
    unsigned* pw1  = (unsigned*)(ws + 1024);
    unsigned* pw2  = (unsigned*)(ws + 5120);
    unsigned* pw3  = (unsigned*)(ws + 7168);
    unsigned* pw4  = (unsigned*)(ws + 9216);
    size_t actoff  = (size_t)1 << 20;
    _Float16* h1 = (_Float16*)(ws + actoff);
    _Float16* h2 = (_Float16*)(ws + actoff + (size_t)nrows * 64 * sizeof(_Float16));
    _Float16* h3 = (_Float16*)(ws + actoff);

    hipMemsetAsync(maxbits, 0, 16, stream);
    prep_weights<<<1, 256, 0, stream>>>(W1, W2, W3, W4, wsc1, wsc2, wsc3, wsc4,
                                        pw1, pw2, pw3, pw4);
    absmax_f32<<<2048, 256, 0, stream>>>(x, (long long)in_sizes[0] / 4, &maxbits[0]);

    constexpr int MT = 4;
    int rpb  = 8 * MT * 16;
    int grid = (nrows + rpb - 1) / rpb;
    qlinear_relu<16, 64, true,  MT><<<grid, 256, 0, stream>>>(x,  b1, wsc1, pw1, &maxbits[0], &maxbits[1], h1, nrows);
    qlinear_relu<64, 32, false, MT><<<grid, 256, 0, stream>>>(h1, b2, wsc2, pw2, &maxbits[1], &maxbits[2], h2, nrows);
    qlinear_relu<32, 32, false, MT><<<grid, 256, 0, stream>>>(h2, b3, wsc3, pw3, &maxbits[2], &maxbits[3], h3, nrows);
    qlinear4_softmax<MT><<<grid, 256, 0, stream>>>(h3, b4, wsc4, pw4, &maxbits[3], (float*)d_out, nrows);
}
